// Block_6760278524294
// MI455X (gfx1250) — compile-verified, matrix operand order checked by reading, not run on hardware
//
#include <hip/hip_runtime.h>
#include <hip/hip_bf16.h>

// ---------------------------------------------------------------------------
// Transformer block for MI455X (gfx1250, wave32, WMMA + TDM).
// GEMMs: v_wmma_f32_16x16x32_bf16 with bf16x2 split precision
// (A = Ahi + Alo, B = Bhi + Blo; accumulate AhiBhi + AhiBlo + AloBhi in f32).
// Tile staging: TENSOR_LOAD_TO_LDS (Tensor Data Mover), double-buffered,
// overlapped with WMMA compute; completion via s_wait_tensorcnt.
// ---------------------------------------------------------------------------

typedef __bf16 bf16_t;
typedef bf16_t v16bf __attribute__((ext_vector_type(16)));
typedef float  v8f   __attribute__((ext_vector_type(8)));
typedef unsigned int u32x4 __attribute__((ext_vector_type(4)));
typedef int          i32x8 __attribute__((ext_vector_type(8)));
typedef int          i32x4 __attribute__((ext_vector_type(4)));

#define Bsz 4
#define Ssz 1024
#define Dsz 1024
#define Hsz 16
#define DHsz 64

#if __has_builtin(__builtin_amdgcn_tensor_load_to_lds)
#define HAVE_TDM 1
#else
#define HAVE_TDM 0
#endif

__device__ __forceinline__ void split_f32(float x, unsigned short& hi, unsigned short& lo) {
    unsigned u = __float_as_uint(x);
    hi = (unsigned short)(u >> 16);                       // truncated hi: exact bf16
    float fh = __uint_as_float(u & 0xFFFF0000u);
    float r  = x - fh;                                    // exact residual
    unsigned ur = __float_as_uint(r);
    lo = (unsigned short)((ur + 0x7FFFu + ((ur >> 16) & 1u)) >> 16);  // RNE bf16
}

// Build a 16x32 bf16 fragment from LDS (ushort), A-layout:
// lane = m(0..15) + 16*half; VGPR v holds K = grp*16 + half*8 + vv*2 (+0,+1).
// For B operands we use the symmetric assumption B-layout(B) == A-layout(B^T),
// i.e. lane indexes the column n and k runs along the row of the (transposed-
// staged) LDS tile, so the same builder serves both operands.
__device__ __forceinline__ v16bf frag_lds(const unsigned short* p, int rs, int lane) {
    int m = lane & 15;
    const unsigned short* row = p + m * rs + ((lane >> 4) << 3); // half*8
    v16bf f;
#pragma unroll
    for (int v = 0; v < 8; ++v) {
        int k = ((v >> 2) << 4) + ((v & 3) << 1);
        f[2 * v]     = __builtin_bit_cast(bf16_t, row[k]);
        f[2 * v + 1] = __builtin_bit_cast(bf16_t, row[k + 1]);
    }
    return f;
}

__device__ __forceinline__ v8f wmma_bf16(v16bf a, v16bf b, v8f c) {
    return __builtin_amdgcn_wmma_f32_16x16x32_bf16(false, a, false, b, (short)0, c, false, false);
}

// ---------------------------------------------------------------------------
// TDM 2D tile load: global (row-major, row_stride elements of f32) -> LDS,
// tile_h rows x tile_w f32 columns, packed contiguously row-major in LDS.
// D# packing per CDNA5 ISA sec. 8.3/8.4:
//  group0: [1:0]=count=1 | [63:32]=lds_addr | [120:64]=global_addr | [127:126]=type=2
//  group1: [17:16]=data_size(2 -> 4B) | tensor_dim0/1 | tile_dim0/1 (tile_dim2=0)
//          | tensor_dim0_stride (48b). tensor dims == tile dims (tile fully
//          in-bounds by construction, OOB logic never triggers).
//  groups 2/3: benign (tensor_dim2=1, rest 0) -- tile_dim2=0 marks them unused.
// This toolchain exposes the 6-arg builtin (clang-23 / therock-10.0 form):
// (u32x4 g0, i32x8 g1, i32x4 g2, i32x4 g3, i32x8 extra, i32 cpol); the extra
// vector is zero-filled (probe-verified pattern).
// ---------------------------------------------------------------------------
__device__ __forceinline__ void tdm_load_2d(const float* gsrc, void* ldst,
                                            unsigned tile_w, unsigned tile_h,
                                            unsigned long long row_stride) {
#if HAVE_TDM
    unsigned long long ga = (unsigned long long)(uintptr_t)gsrc;
    u32x4 g0;
    g0[0] = 1u;                                              // count=1, user D#
    g0[1] = (unsigned)(uintptr_t)ldst;                       // LDS byte offset
    g0[2] = (unsigned)ga;                                    // global_addr[31:0]
    g0[3] = (unsigned)((ga >> 32) & 0x01FFFFFFu) | (2u << 30); // [56:32] | type=2
    i32x8 g1;
    g1[0] = (int)(2u << 16);                                 // data_size=4B
    g1[1] = (int)((tile_w & 0xFFFFu) << 16);                 // tensor_dim0 lo16
    g1[2] = (int)((tile_w >> 16) | ((tile_h & 0xFFFFu) << 16)); // d0 hi / d1 lo
    g1[3] = (int)((tile_h >> 16) | (tile_w << 16));          // d1 hi | tile_dim0
    g1[4] = (int)tile_h;                                     // tile_dim1, tile_dim2=0
    unsigned long long sb = row_stride;                      // in elements
    g1[5] = (int)(sb & 0xFFFFFFFFu);                         // dim0_stride lo
    g1[6] = (int)((sb >> 32) & 0xFFFFu);                     // dim0_stride hi
    g1[7] = 0;                                               // dim1_stride (2D)
    i32x4 g2, g3;
    g2[0] = 1; g2[1] = 0; g2[2] = 0; g2[3] = 0;
    g3[0] = 0; g3[1] = 0; g3[2] = 0; g3[3] = 0;
    i32x8 gx;                                                // uncharacterized extra
    gx[0] = 0; gx[1] = 0; gx[2] = 0; gx[3] = 0;
    gx[4] = 0; gx[5] = 0; gx[6] = 0; gx[7] = 0;
    __builtin_amdgcn_tensor_load_to_lds(g0, g1, g2, g3, gx, 0);
#else
    // Fallback: synchronous copy by the issuing wave.
    int lane = threadIdx.x & 31;
    float* d = (float*)ldst;
    for (unsigned i = lane; i < tile_w * tile_h; i += 32) {
        unsigned r = i / tile_w, c = i % tile_w;
        d[i] = gsrc[(size_t)r * row_stride + c];
    }
#endif
}

__device__ __forceinline__ void tdm_wait_pending2() {
#if HAVE_TDM
    __builtin_amdgcn_s_wait_tensorcnt(2);
#endif
}
__device__ __forceinline__ void tdm_wait_all() {
#if HAVE_TDM
    __builtin_amdgcn_s_wait_tensorcnt(0);
#endif
}

// ---------------------------------------------------------------------------
// GEMM: C[M,N] = A[M,K] @ B[K,N] + bias (+ optional ReLU). M,N,K % 128 == 0.
// 256 threads = 8 waves (4x2), each wave computes 32x64 of the 128x128 tile.
// fp32 tiles arrive by TDM DMA (double-buffered, one k-step ahead); threads
// then split fp32 -> bf16 hi/lo into the WMMA staging arrays (B transposed).
// ---------------------------------------------------------------------------
__global__ __launch_bounds__(256) void gemm_wmma_kernel(
    const float* __restrict__ A, const float* __restrict__ B,
    const float* __restrict__ bias, float* __restrict__ C,
    int M, int N, int K, int relu)
{
    __shared__ float As32[2][128][32];                    // TDM landing (A)
    __shared__ float Bs32[2][32][128];                    // TDM landing (B)
    __shared__ unsigned short Ah[128][34], Al[128][34];
    __shared__ unsigned short Bh[128][34], Bl[128][34];   // staged transposed: [n][k]

    const int tid  = threadIdx.x;
    const int lane = tid & 31;
    const int wave = tid >> 5;
    const int wm   = wave >> 1;          // 0..3
    const int wn   = wave & 1;           // 0..1
    const int brow = blockIdx.y * 128;
    const int bcol = blockIdx.x * 128;

    v8f acc[2][4];
#pragma unroll
    for (int i = 0; i < 2; ++i)
#pragma unroll
        for (int j = 0; j < 4; ++j) acc[i][j] = (v8f)(0.0f);

    const int nsteps = K >> 5;
    if (wave == 0) {                     // prologue: DMA first tiles into buf0
        tdm_load_2d(A + (size_t)brow * K, &As32[0][0][0], 32, 128, (unsigned long long)K);
        tdm_load_2d(B + bcol,             &Bs32[0][0][0], 128, 32, (unsigned long long)N);
    }

    for (int s = 0; s < nsteps; ++s) {
        const int cur = s & 1;
        const int k0  = s << 5;
        if (wave == 0) {
            if (s + 1 < nsteps) {        // DMA next k-step into the other buffer
                tdm_load_2d(A + (size_t)brow * K + (k0 + 32), &As32[1 - cur][0][0],
                            32, 128, (unsigned long long)K);
                tdm_load_2d(B + (size_t)(k0 + 32) * N + bcol, &Bs32[1 - cur][0][0],
                            128, 32, (unsigned long long)N);
                tdm_wait_pending2();     // in-order: current pair complete
            } else {
                tdm_wait_all();
            }
        }
        __syncthreads();                 // fp32 tiles for step s are visible

        // split fp32 -> bf16 hi/lo staging (each element split exactly once)
#pragma unroll
        for (int i = 0; i < 16; ++i) {
            int e = tid * 16 + i;
            int r = e >> 5, c = e & 31;
            unsigned short hi, lo; split_f32(As32[cur][r][c], hi, lo);
            Ah[r][c] = hi; Al[r][c] = lo;
        }
#pragma unroll
        for (int i = 0; i < 16; ++i) {
            int e = tid * 16 + i;
            int kk = e >> 7, n = e & 127;
            unsigned short hi, lo; split_f32(Bs32[cur][kk][n], hi, lo);
            Bh[n][kk] = hi; Bl[n][kk] = lo;
        }
        __syncthreads();                 // bf16 staging ready

        v16bf afh[2], afl[2], bfh[4], bfl[4];
#pragma unroll
        for (int t = 0; t < 2; ++t) {
            afh[t] = frag_lds(&Ah[wm * 32 + t * 16][0], 34, lane);
            afl[t] = frag_lds(&Al[wm * 32 + t * 16][0], 34, lane);
        }
#pragma unroll
        for (int t = 0; t < 4; ++t) {
            bfh[t] = frag_lds(&Bh[wn * 64 + t * 16][0], 34, lane);
            bfl[t] = frag_lds(&Bl[wn * 64 + t * 16][0], 34, lane);
        }
#pragma unroll
        for (int mt = 0; mt < 2; ++mt)
#pragma unroll
            for (int nt = 0; nt < 4; ++nt) {
                acc[mt][nt] = wmma_bf16(afh[mt], bfh[nt], acc[mt][nt]);
                acc[mt][nt] = wmma_bf16(afh[mt], bfl[nt], acc[mt][nt]);
                acc[mt][nt] = wmma_bf16(afl[mt], bfh[nt], acc[mt][nt]);
            }
        // next iteration's top barrier orders these fragment reads before the
        // bf16 staging arrays are overwritten.
    }

    // Epilogue. C layout: VGPR r, lane -> row = r + 8*(lane>>4), col = lane&15.
    const int half = lane >> 4;
#pragma unroll
    for (int mt = 0; mt < 2; ++mt) {
        int rbase = brow + wm * 32 + mt * 16 + half * 8;
#pragma unroll
        for (int nt = 0; nt < 4; ++nt) {
            int col = bcol + wn * 64 + nt * 16 + (lane & 15);
            float bv = bias[col];
#pragma unroll
            for (int r = 0; r < 8; ++r) {
                float v = acc[mt][nt][r] + bv;
                if (relu) v = fmaxf(v, 0.0f);
                C[(size_t)(rbase + r) * N + col] = v;
            }
        }
    }
}

// ---------------------------------------------------------------------------
// Causal flash attention. Grid: (S/128, H, B); 256 threads = 8 waves,
// each wave owns 16 query rows. Streams key blocks of 32, WMMA for QK^T & PV.
// qkv layout: [B,S,3D] with q at +0, k at +D, v at +2D; per head offset h*64.
// ---------------------------------------------------------------------------
__global__ __launch_bounds__(256) void attn_kernel(
    const float* __restrict__ qkv, float* __restrict__ aout)
{
    __shared__ unsigned short Kh[32][66], Kl[32][66];     // [key][dh]
    __shared__ unsigned short Vh[64][34], Vl[64][34];     // transposed: [dh][key]
    __shared__ unsigned short Ph[8][16][34], Pl[8][16][34];

    const int tid  = threadIdx.x;
    const int lane = tid & 31;
    const int wave = tid >> 5;
    const int qb   = blockIdx.x;
    const int head = blockIdx.y;
    const int b    = blockIdx.z;
    const int m    = lane & 15;
    const int half = lane >> 4;
    const int rowbase = qb * 128 + wave * 16;

    // Q fragments in registers (2 K-steps x hi/lo)
    v16bf qfh[2], qfl[2];
    {
        const float* qp = qkv + (size_t)(b * Ssz + rowbase + m) * (3 * Dsz) + head * DHsz;
#pragma unroll
        for (int ks = 0; ks < 2; ++ks)
#pragma unroll
            for (int v = 0; v < 8; ++v) {
                int k = ks * 32 + ((v >> 2) << 4) + (half << 3) + ((v & 3) << 1);
                unsigned short h0, l0, h1, l1;
                split_f32(qp[k], h0, l0);
                split_f32(qp[k + 1], h1, l1);
                qfh[ks][2 * v] = __builtin_bit_cast(bf16_t, h0);
                qfh[ks][2 * v + 1] = __builtin_bit_cast(bf16_t, h1);
                qfl[ks][2 * v] = __builtin_bit_cast(bf16_t, l0);
                qfl[ks][2 * v + 1] = __builtin_bit_cast(bf16_t, l1);
            }
    }

    v8f o[4];
#pragma unroll
    for (int t = 0; t < 4; ++t) o[t] = (v8f)(0.0f);
    float mrun[8], lrun[8];
#pragma unroll
    for (int r = 0; r < 8; ++r) { mrun[r] = -3.0e38f; lrun[r] = 0.0f; }

    const int nkb = qb * 4 + 4;   // causal: keys up to (qb*128 + 127)
    for (int kb = 0; kb < nkb; ++kb) {
        __syncthreads();
        // Stage K,V blocks (32 keys x 64 dh), 8 elems each per thread
#pragma unroll
        for (int i = 0; i < 8; ++i) {
            int e = tid * 8 + i;
            int kk = e >> 6, dh = e & 63;
            size_t base = (size_t)(b * Ssz + kb * 32 + kk) * (3 * Dsz) + head * DHsz + dh;
            unsigned short hi, lo;
            split_f32(qkv[base + Dsz], hi, lo);        // K
            Kh[kk][dh] = hi; Kl[kk][dh] = lo;
            split_f32(qkv[base + 2 * Dsz], hi, lo);    // V (stored transposed)
            Vh[dh][kk] = hi; Vl[dh][kk] = lo;
        }
        __syncthreads();

        // scores: 16x32 per wave = 2 n-tiles, K depth 64 = 2 k-steps
        v8f s[2]; s[0] = (v8f)(0.0f); s[1] = (v8f)(0.0f);
#pragma unroll
        for (int nt = 0; nt < 2; ++nt)
#pragma unroll
            for (int ks = 0; ks < 2; ++ks) {
                v16bf kh = frag_lds(&Kh[nt * 16][ks * 32], 66, lane);
                v16bf kl = frag_lds(&Kl[nt * 16][ks * 32], 66, lane);
                s[nt] = wmma_bf16(qfh[ks], kh, s[nt]);
                s[nt] = wmma_bf16(qfh[ks], kl, s[nt]);
                s[nt] = wmma_bf16(qfl[ks], kh, s[nt]);
            }
        // scale + causal mask (masked entries are exactly -1e9 as in reference)
#pragma unroll
        for (int nt = 0; nt < 2; ++nt) {
            int colg = kb * 32 + nt * 16 + (lane & 15);
#pragma unroll
            for (int r = 0; r < 8; ++r) {
                int rowg = rowbase + r + half * 8;
                float v = s[nt][r] * 0.125f;
                s[nt][r] = (colg <= rowg) ? v : -1.0e9f;
            }
        }
        // online softmax (row reductions across the 16-lane half-groups)
        float pc[8];
#pragma unroll
        for (int r = 0; r < 8; ++r) {
            float mx = fmaxf(s[0][r], s[1][r]);
#pragma unroll
            for (int off = 8; off > 0; off >>= 1) mx = fmaxf(mx, __shfl_xor(mx, off, 32));
            float nm = fmaxf(mrun[r], mx);
            float c = __expf(mrun[r] - nm);
            float p0 = __expf(s[0][r] - nm);
            float p1 = __expf(s[1][r] - nm);
            s[0][r] = p0; s[1][r] = p1;
            float ps = p0 + p1;
#pragma unroll
            for (int off = 8; off > 0; off >>= 1) ps += __shfl_xor(ps, off, 32);
            lrun[r] = lrun[r] * c + ps;
            mrun[r] = nm;
            pc[r] = c;
        }
#pragma unroll
        for (int t = 0; t < 4; ++t)
#pragma unroll
            for (int r = 0; r < 8; ++r) o[t][r] *= pc[r];

        // P (C layout) -> wave-private LDS -> A fragments (hi/lo)
#pragma unroll
        for (int nt = 0; nt < 2; ++nt)
#pragma unroll
            for (int r = 0; r < 8; ++r) {
                unsigned short hi, lo; split_f32(s[nt][r], hi, lo);
                Ph[wave][r + half * 8][nt * 16 + (lane & 15)] = hi;
                Pl[wave][r + half * 8][nt * 16 + (lane & 15)] = lo;
            }
        v16bf pfh = frag_lds(&Ph[wave][0][0], 34, lane);
        v16bf pfl = frag_lds(&Pl[wave][0][0], 34, lane);
        // O += P @ V  (K depth 32 = this key block)
#pragma unroll
        for (int t = 0; t < 4; ++t) {
            v16bf vh = frag_lds(&Vh[t * 16][0], 34, lane);
            v16bf vl = frag_lds(&Vl[t * 16][0], 34, lane);
            o[t] = wmma_bf16(pfh, vh, o[t]);
            o[t] = wmma_bf16(pfh, vl, o[t]);
            o[t] = wmma_bf16(pfl, vh, o[t]);
        }
    }

    // normalize and store: a[b, row, head*64 + dh] (un-transposed head merge)
#pragma unroll
    for (int t = 0; t < 4; ++t) {
        int dh = t * 16 + (lane & 15);
#pragma unroll
        for (int r = 0; r < 8; ++r) {
            int rowg = rowbase + r + half * 8;
            aout[(size_t)(b * Ssz + rowg) * Dsz + head * DHsz + dh] = o[t][r] / lrun[r];
        }
    }
}

// ---------------------------------------------------------------------------
// Residual add + the reference's LayerNorm: g*(x-mean)/sqrt(std+eps)+b,
// std = sqrt(var(ddof=1)). One 256-thread block per row (D=1024).
// ---------------------------------------------------------------------------
__global__ __launch_bounds__(256) void add_ln_kernel(
    const float* __restrict__ x, const float* __restrict__ y,
    const float* __restrict__ g, const float* __restrict__ bta,
    float* __restrict__ out)
{
    __shared__ float reds[8], redq[8], bcast[2];
    const int tid = threadIdx.x;
    const int lane = tid & 31, wave = tid >> 5;
    const size_t row = blockIdx.x;
    const float* px = x + row * Dsz;
    const float* py = y + row * Dsz;

    float vals[4];
    float s = 0.f, ss = 0.f;
#pragma unroll
    for (int i = 0; i < 4; ++i) {
        float v = px[tid + i * 256] + py[tid + i * 256];
        vals[i] = v; s += v; ss += v * v;
    }
#pragma unroll
    for (int off = 16; off > 0; off >>= 1) {
        s  += __shfl_xor(s, off, 32);
        ss += __shfl_xor(ss, off, 32);
    }
    if (lane == 0) { reds[wave] = s; redq[wave] = ss; }
    __syncthreads();
    if (tid == 0) {
        float S = 0.f, Q = 0.f;
#pragma unroll
        for (int w = 0; w < 8; ++w) { S += reds[w]; Q += redq[w]; }
        float mean = S * (1.0f / Dsz);
        float var  = fmaxf((Q - (float)Dsz * mean * mean) * (1.0f / (Dsz - 1)), 0.0f);
        float stdv = __builtin_sqrtf(var);
        bcast[0] = mean;
        bcast[1] = 1.0f / __builtin_sqrtf(stdv + 1e-5f);
    }
    __syncthreads();
    float mean = bcast[0], inv = bcast[1];
#pragma unroll
    for (int i = 0; i < 4; ++i) {
        int c = tid + i * 256;
        out[row * Dsz + c] = g[c] * (vals[i] - mean) * inv + bta[c];
    }
}

// ---------------------------------------------------------------------------
// Launch: qkv GEMM -> attention -> proj GEMM -> LN1 -> FC(ReLU) -> proj -> LN2
// Workspace (floats): qkv[12.58M] | a[4.19M] | h[4.19M] | fc[16.78M]
// p1 and m alias the (dead-by-then) qkv region. Total ~151 MB.
// ---------------------------------------------------------------------------
extern "C" void kernel_launch(void* const* d_in, const int* in_sizes, int n_in,
                              void* d_out, int out_size, void* d_ws, size_t ws_size,
                              hipStream_t stream) {
    const float* x           = (const float*)d_in[0];
    const float* w_qkv       = (const float*)d_in[1];
    const float* b_qkv       = (const float*)d_in[2];
    const float* w_attn_proj = (const float*)d_in[3];
    const float* b_attn_proj = (const float*)d_in[4];
    const float* ln1_g       = (const float*)d_in[5];
    const float* ln1_b       = (const float*)d_in[6];
    const float* w_fc        = (const float*)d_in[7];
    const float* b_fc        = (const float*)d_in[8];
    const float* w_mlp_proj  = (const float*)d_in[9];
    const float* b_mlp_proj  = (const float*)d_in[10];
    const float* ln2_g       = (const float*)d_in[11];
    const float* ln2_b       = (const float*)d_in[12];
    float* out = (float*)d_out;

    float* ws  = (float*)d_ws;
    const size_t MS = (size_t)Bsz * Ssz;           // 4096 rows
    float* qkv = ws;                               // [4096, 3072]
    float* a   = ws + MS * 3 * Dsz;                // [4096, 1024]
    float* h   = a + MS * Dsz;                     // [4096, 1024]
    float* fc  = h + MS * Dsz;                     // [4096, 4096]
    float* p1  = ws;                               // alias qkv (dead after attn)
    float* m2  = ws + MS * Dsz;                    // alias qkv region, disjoint from p1

    dim3 blk(256);
    // 1) qkv = x @ w_qkv + b_qkv
    gemm_wmma_kernel<<<dim3(3 * Dsz / 128, MS / 128), blk, 0, stream>>>(
        x, w_qkv, b_qkv, qkv, (int)MS, 3 * Dsz, Dsz, 0);
    // 2) causal attention -> a
    attn_kernel<<<dim3(Ssz / 128, Hsz, Bsz), blk, 0, stream>>>(qkv, a);
    // 3) p1 = a @ w_attn_proj + b
    gemm_wmma_kernel<<<dim3(Dsz / 128, MS / 128), blk, 0, stream>>>(
        a, w_attn_proj, b_attn_proj, p1, (int)MS, Dsz, Dsz, 0);
    // 4) h = LN1(x + p1)
    add_ln_kernel<<<dim3((unsigned)MS), blk, 0, stream>>>(x, p1, ln1_g, ln1_b, h);
    // 5) fc = relu(h @ w_fc + b_fc)
    gemm_wmma_kernel<<<dim3(4 * Dsz / 128, MS / 128), blk, 0, stream>>>(
        h, w_fc, b_fc, fc, (int)MS, 4 * Dsz, Dsz, 1);
    // 6) m2 = fc @ w_mlp_proj + b
    gemm_wmma_kernel<<<dim3(Dsz / 128, MS / 128), blk, 0, stream>>>(
        fc, w_mlp_proj, b_mlp_proj, m2, (int)MS, Dsz, 4 * Dsz, 0);
    // 7) out = LN2(h + m2)
    add_ln_kernel<<<dim3((unsigned)MS), blk, 0, stream>>>(h, m2, ln2_g, ln2_b, out);
}
